// EdgePoolingHack_62843961475620
// MI455X (gfx1250) — matrix operand order repaired and checked by reading, not compiled, not run on hardware
//
#include <hip/hip_runtime.h>
#include <stdint.h>

// ---------------------------------------------------------------------------
// Types for CDNA5 WMMA
// ---------------------------------------------------------------------------
typedef __attribute__((ext_vector_type(16))) __bf16 v16bf;
typedef __attribute__((ext_vector_type(8)))  float  v8f;

union FragB16 { uint4 q[2]; v16bf v; };

__device__ __forceinline__ unsigned short f2bf(float f) {
  unsigned int u = __builtin_bit_cast(unsigned int, f);
  u += 0x7FFFu + ((u >> 16) & 1u);      // round-to-nearest-even
  return (unsigned short)(u >> 16);
}

// CDNA5 async DMA: global -> LDS, 16B per lane, tracked by ASYNCcnt.
// lds_off: byte offset within the wave's LDS (low 32 bits of generic ptr).
__device__ __forceinline__ void async_load_b128(uint32_t lds_off, const void* gaddr) {
  asm volatile("global_load_async_to_lds_b128 %0, %1, off"
               :: "v"(lds_off), "v"((unsigned long long)(uintptr_t)gaddr)
               : "memory");
}
__device__ __forceinline__ void wait_async0() {
  asm volatile("s_wait_asynccnt 0x0" ::: "memory");
}

// ---------------------------------------------------------------------------
// Conversions
// ---------------------------------------------------------------------------
__global__ void k_cvt_bf16(const float* __restrict__ in,
                           unsigned short* __restrict__ out, int n) {
  int i = blockIdx.x * 256 + threadIdx.x;
  if (i < n) out[i] = f2bf(in[i]);
}

// in: f32 [K][Nn] row-major, out: bf16 [Nn][K]
__global__ void k_transpose_bf16(const float* __restrict__ in,
                                 unsigned short* __restrict__ out,
                                 int K, int Nn) {
  int i = blockIdx.x * 256 + threadIdx.x;
  if (i < K * Nn) {
    int k = i / Nn, n = i % Nn;
    out[n * K + k] = f2bf(in[i]);
  }
}

__global__ void k_scores_init(float* __restrict__ s,
                              const float* __restrict__ b2, int E) {
  int i = blockIdx.x * 256 + threadIdx.x;
  if (i < E) s[i] = b2[0] + 0.5f;
}

__global__ void k_fill_int(int* __restrict__ p, int val, int n) {
  int i = blockIdx.x * 256 + threadIdx.x;
  if (i < n) p[i] = val;
}

// ---------------------------------------------------------------------------
// Edge MLP: scores = relu(concat(x[src],x[dst]) @ W1 + b1) @ W2  (+b2+0.5 via init)
// Block: 256 threads / 8 waves, tile M=128 edges, N=256, K=256 all-in-LDS.
// Staging via global_load_async_to_lds_b128 (ASYNCcnt DMA, no VGPR round-trip).
// ---------------------------------------------------------------------------
#define A_STRIDE 264   // bf16 units per A row (256 + 8 pad)
#define W_STRIDE 264   // bf16 units per W row (256 + 8 pad)
// smem bytes: 128*264*2 + 256*264*2 = 67584 + 135168 = 202752

__global__ __launch_bounds__(256) void k_edge_mlp(
    const unsigned short* __restrict__ xh,   // [N][128] bf16
    const unsigned short* __restrict__ W1t,  // [256][256] bf16 (out-major)
    const float* __restrict__ b1,            // [256]
    const float* __restrict__ W2,            // [256]
    const int* __restrict__ src, const int* __restrict__ dst,
    float* __restrict__ scores) {
  extern __shared__ unsigned short smem[];
  unsigned short* sA = smem;                       // [128][A_STRIDE]
  unsigned short* sW = smem + 128 * A_STRIDE;      // [256][W_STRIDE]
  const int tid = threadIdx.x;
  const int w = tid >> 5, l = tid & 31;
  const int e0 = blockIdx.x * 128;

  // stage W1t: thread t DMAs its row (256 bf16 = 32 x 16B)
  {
    const unsigned short* g = W1t + (size_t)tid * 256;
    uint32_t d = (uint32_t)(uintptr_t)(sW + tid * W_STRIDE);
#pragma unroll
    for (int q = 0; q < 32; ++q)
      async_load_b128(d + q * 16, g + q * 8);
  }
  // stage gathered A tile: wave w stages rows w*16..w*16+15
  {
#pragma unroll
    for (int r = 0; r < 16; ++r) {
      int row = w * 16 + r;
      int e = e0 + row;
      int node = (l < 16) ? src[e] : dst[e];
      const unsigned short* g = xh + (size_t)node * 128 + (l & 15) * 8;
      uint32_t d = (uint32_t)(uintptr_t)(sA + row * A_STRIDE + (l >> 4) * 128 + (l & 15) * 8);
      async_load_b128(d, g);
    }
  }
  wait_async0();
  __syncthreads();

  const int mq = w >> 1;        // 0..3 : M quadrant (32 edges)
  const int nh = w & 1;         // 0..1 : N half (128 cols)
  const int mbase = mq * 32;
  const int nbase = nh * 128;
  const int lm = l & 15;
  const int lh = l >> 4;

  v8f acc[2][8];
  const v8f vzero = {0.f, 0.f, 0.f, 0.f, 0.f, 0.f, 0.f, 0.f};
#pragma unroll
  for (int a = 0; a < 2; ++a)
#pragma unroll
    for (int b = 0; b < 8; ++b) acc[a][b] = vzero;

#pragma unroll
  for (int kt = 0; kt < 8; ++kt) {
    FragB16 fa[2];
#pragma unroll
    for (int mt = 0; mt < 2; ++mt) {
      int m = mbase + mt * 16 + lm;
      int ka = kt * 32 + lh * 16;                // A frag: K contiguous 16h..16h+15
      fa[mt].q[0] = *((const uint4*)(sA + m * A_STRIDE + ka));
      fa[mt].q[1] = *((const uint4*)(sA + m * A_STRIDE + ka + 8));
    }
    // double-buffered B fragments: issue loads one nt ahead of consumption
    FragB16 fb[2];
    {
      int n = nbase + lm;
      int kb = kt * 32 + lh * 16;                // B frag: lanes=col, K contiguous
      fb[0].q[0] = *((const uint4*)(sW + n * W_STRIDE + kb));
      fb[0].q[1] = *((const uint4*)(sW + n * W_STRIDE + kb + 8));
    }
#pragma unroll
    for (int nt = 0; nt < 8; ++nt) {
      int cur = nt & 1, nxt = cur ^ 1;
      if (nt < 7) {
        int n = nbase + (nt + 1) * 16 + lm;
        int kb = kt * 32 + lh * 16;
        fb[nxt].q[0] = *((const uint4*)(sW + n * W_STRIDE + kb));
        fb[nxt].q[1] = *((const uint4*)(sW + n * W_STRIDE + kb + 8));
      }
      acc[0][nt] = __builtin_amdgcn_wmma_f32_16x16x32_bf16(
          false, fa[0].v, false, fb[cur].v, (short)0, acc[0][nt], false, false);
      acc[1][nt] = __builtin_amdgcn_wmma_f32_16x16x32_bf16(
          false, fa[1].v, false, fb[cur].v, (short)0, acc[1][nt], false, false);
    }
  }

  // fused layer 2: bias + relu + dot with W2, reduce across 16-lane halves
  float ps[2][8];
#pragma unroll
  for (int mt = 0; mt < 2; ++mt)
#pragma unroll
    for (int v = 0; v < 8; ++v) ps[mt][v] = 0.f;

#pragma unroll
  for (int nt = 0; nt < 8; ++nt) {
    int n = nbase + nt * 16 + lm;
    float bn = b1[n], wn = W2[n];
#pragma unroll
    for (int mt = 0; mt < 2; ++mt)
#pragma unroll
      for (int v = 0; v < 8; ++v) {
        float h = acc[mt][nt][v] + bn;
        h = h > 0.f ? h : 0.f;
        ps[mt][v] += h * wn;
      }
  }
#pragma unroll
  for (int mt = 0; mt < 2; ++mt)
#pragma unroll
    for (int v = 0; v < 8; ++v) {
      float s = ps[mt][v];
      s += __shfl_xor(s, 1, 32);
      s += __shfl_xor(s, 2, 32);
      s += __shfl_xor(s, 4, 32);
      s += __shfl_xor(s, 8, 32);
      ps[mt][v] = s;
    }
  if (lm == 0) {
#pragma unroll
    for (int mt = 0; mt < 2; ++mt)
#pragma unroll
      for (int v = 0; v < 8; ++v) {
        int e = e0 + mbase + mt * 16 + v + lh * 8;   // D layout: lanes>=16 -> M=v+8
        atomicAdd(&scores[e], ps[mt][v]);
      }
  }
}

// ---------------------------------------------------------------------------
// Generic [M,128] @ [128,128] + bias, bf16 in / f32 out.  Tile M=128.
// ---------------------------------------------------------------------------
#define NA_STRIDE 136   // 128 + 8 pad
// smem bytes: 2 * 128*136*2 = 69632

__global__ __launch_bounds__(256) void k_gemm128(
    const unsigned short* __restrict__ A,   // [M][128] bf16
    const unsigned short* __restrict__ Wt,  // [128][128] bf16 (out-major)
    const float* __restrict__ bias,         // [128]
    float* __restrict__ out) {              // [M][128]
  extern __shared__ unsigned short smem[];
  unsigned short* sA = smem;                      // [128][136]
  unsigned short* sW = smem + 128 * NA_STRIDE;    // [128][136]
  const int tid = threadIdx.x;
  const int w = tid >> 5, l = tid & 31;
  const size_t r0 = (size_t)blockIdx.x * 128;

  {  // stage Wt via async DMA
    int row = tid >> 1, half = tid & 1;
    const unsigned short* g = Wt + (size_t)row * 128 + half * 64;
    uint32_t d = (uint32_t)(uintptr_t)(sW + row * NA_STRIDE + half * 64);
#pragma unroll
    for (int q = 0; q < 8; ++q)
      async_load_b128(d + q * 16, g + q * 8);
  }
  {  // stage A via async DMA: wave w rows w*16..+15, two rows per iter
#pragma unroll
    for (int i = 0; i < 8; ++i) {
      int row = w * 16 + i * 2 + (l >> 4);
      const unsigned short* g = A + (r0 + row) * 128 + (l & 15) * 8;
      uint32_t d = (uint32_t)(uintptr_t)(sA + row * NA_STRIDE + (l & 15) * 8);
      async_load_b128(d, g);
    }
  }
  wait_async0();
  __syncthreads();

  const int mq = w >> 1, nh = w & 1;
  const int mbase = mq * 32, nbase = nh * 64;
  const int lm = l & 15, lh = l >> 4;

  v8f acc[2][4];
  const v8f vzero = {0.f, 0.f, 0.f, 0.f, 0.f, 0.f, 0.f, 0.f};
#pragma unroll
  for (int a = 0; a < 2; ++a)
#pragma unroll
    for (int b = 0; b < 4; ++b) acc[a][b] = vzero;

#pragma unroll
  for (int kt = 0; kt < 4; ++kt) {
    FragB16 fa[2];
#pragma unroll
    for (int mt = 0; mt < 2; ++mt) {
      int m = mbase + mt * 16 + lm;
      int ka = kt * 32 + lh * 16;
      fa[mt].q[0] = *((const uint4*)(sA + m * NA_STRIDE + ka));
      fa[mt].q[1] = *((const uint4*)(sA + m * NA_STRIDE + ka + 8));
    }
    FragB16 fb[2];
    {
      int n = nbase + lm;
      int kb = kt * 32 + lh * 16;
      fb[0].q[0] = *((const uint4*)(sW + n * NA_STRIDE + kb));
      fb[0].q[1] = *((const uint4*)(sW + n * NA_STRIDE + kb + 8));
    }
#pragma unroll
    for (int nt = 0; nt < 4; ++nt) {
      int cur = nt & 1, nxt = cur ^ 1;
      if (nt < 3) {
        int n = nbase + (nt + 1) * 16 + lm;
        int kb = kt * 32 + lh * 16;
        fb[nxt].q[0] = *((const uint4*)(sW + n * NA_STRIDE + kb));
        fb[nxt].q[1] = *((const uint4*)(sW + n * NA_STRIDE + kb + 8));
      }
      acc[0][nt] = __builtin_amdgcn_wmma_f32_16x16x32_bf16(
          false, fa[0].v, false, fb[cur].v, (short)0, acc[0][nt], false, false);
      acc[1][nt] = __builtin_amdgcn_wmma_f32_16x16x32_bf16(
          false, fa[1].v, false, fb[cur].v, (short)0, acc[1][nt], false, false);
    }
  }

#pragma unroll
  for (int mt = 0; mt < 2; ++mt)
#pragma unroll
    for (int nt = 0; nt < 4; ++nt) {
      int n = nbase + nt * 16 + lm;
      float bn = bias[n];
#pragma unroll
      for (int v = 0; v < 8; ++v) {
        int m = mbase + mt * 16 + v + lh * 8;
        out[(r0 + m) * 128 + n] = acc[mt][nt][v] + bn;
      }
    }
}

// ---------------------------------------------------------------------------
// Per-graph argmax over contiguous 4096-edge chunks (first-max tie-break)
// ---------------------------------------------------------------------------
__global__ void k_argmax(const float* __restrict__ scores,
                         const int* __restrict__ src, const int* __restrict__ dst,
                         int* __restrict__ sel_src, int* __restrict__ sel_dst,
                         int epg) {
  __shared__ float ss[256];
  __shared__ int   si[256];
  int g = blockIdx.x, t = threadIdx.x;
  float best = -3.4e38f;
  int bi = 0x7fffffff;
  for (int e = t; e < epg; e += 256) {
    int ge = g * epg + e;
    float s = scores[ge];
    if (s > best) { best = s; bi = ge; }
  }
  ss[t] = best; si[t] = bi;
  __syncthreads();
  for (int o = 128; o > 0; o >>= 1) {
    if (t < o) {
      if (ss[t + o] > ss[t] || (ss[t + o] == ss[t] && si[t + o] < si[t])) {
        ss[t] = ss[t + o]; si[t] = si[t + o];
      }
    }
    __syncthreads();
  }
  if (t == 0) {
    int fi = si[0];
    sel_src[g] = src[fi];
    sel_dst[g] = dst[fi];
  }
}

// ---------------------------------------------------------------------------
// Cluster / mask bookkeeping
// ---------------------------------------------------------------------------
__global__ void k_mark(const int* __restrict__ sel_src, const int* __restrict__ sel_dst,
                       int* __restrict__ flag, int* __restrict__ cluster, int Bg) {
  int i = threadIdx.x;
  if (i < Bg) {
    int a = sel_src[i], b = sel_dst[i];
    flag[a] = 0; flag[b] = 0;
    cluster[a] = i; cluster[b] = i;
  }
}

__global__ void k_remidx(const int* __restrict__ flag, const int* __restrict__ pos,
                         int* __restrict__ rem_idx, int* __restrict__ cluster,
                         int Nn, int Bg) {
  int i = blockIdx.x * 256 + threadIdx.x;
  if (i < Nn && flag[i]) {
    int r = pos[i];
    rem_idx[r] = i;
    cluster[i] = Bg + r;
  }
}

__global__ void k_build_tmp(const float* __restrict__ xm,
                            const int* __restrict__ sel_src, const int* __restrict__ sel_dst,
                            const int* __restrict__ rem_idx,
                            unsigned short* __restrict__ tmpbf,
                            int num_new, int Bg) {
  int i = blockIdx.x * 256 + threadIdx.x;
  if (i >= num_new * 128) return;
  int row = i >> 7, c = i & 127;
  float v;
  if (row < Bg)
    v = xm[(size_t)sel_src[row] * 128 + c] + xm[(size_t)sel_dst[row] * 128 + c];
  else
    v = xm[(size_t)rem_idx[row - Bg] * 128 + c];
  tmpbf[i] = f2bf(v);
}

__global__ void k_new_batch(const int* __restrict__ batch, const int* __restrict__ rem_idx,
                            int* __restrict__ nb, int num_new, int Bg) {
  int i = blockIdx.x * 256 + threadIdx.x;
  if (i < num_new) nb[i] = (i < Bg) ? i : batch[rem_idx[i - Bg]];
}

// ---------------------------------------------------------------------------
// Pair coalesce: keys, bitonic sort, unique-compact
// ---------------------------------------------------------------------------
__global__ void k_keys(const int* __restrict__ src, const int* __restrict__ dst,
                       const int* __restrict__ cluster,
                       unsigned long long* __restrict__ keys, int E) {
  int e = blockIdx.x * 256 + threadIdx.x;
  if (e < E)
    keys[e] = ((unsigned long long)(unsigned)cluster[src[e]] << 32) |
              (unsigned)cluster[dst[e]];
}

__global__ void k_bitonic(unsigned long long* __restrict__ keys, int j, int k) {
  int i = blockIdx.x * 256 + threadIdx.x;
  int ixj = i ^ j;
  if (ixj > i) {
    unsigned long long a = keys[i], b = keys[ixj];
    bool asc = ((i & k) == 0);
    if ((a > b) == asc) { keys[i] = b; keys[ixj] = a; }
  }
}

__global__ void k_uniq(const unsigned long long* __restrict__ keys,
                       int* __restrict__ uf, int E) {
  int i = blockIdx.x * 256 + threadIdx.x;
  if (i < E) uf[i] = (i == 0 || keys[i] != keys[i - 1]) ? 1 : 0;
}

__global__ void k_scatter_unique(const unsigned long long* __restrict__ keys,
                                 const int* __restrict__ uf, const int* __restrict__ upos,
                                 int* __restrict__ e0, int* __restrict__ e1, int E) {
  int i = blockIdx.x * 256 + threadIdx.x;
  if (i < E && uf[i]) {
    int p = upos[i];
    unsigned long long k = keys[i];
    e0[p] = (int)(k >> 32);
    e1[p] = (int)(k & 0xffffffffu);
  }
}

// ---------------------------------------------------------------------------
// Exclusive scan (1024 items / block; <=1024 blocks)
// ---------------------------------------------------------------------------
__global__ void k_scan_block(const int* __restrict__ in, int* __restrict__ out,
                             int* __restrict__ bsums, int n) {
  __shared__ int s[256];
  int t = threadIdx.x;
  int base = blockIdx.x * 1024 + t * 4;
  int v0 = 0, v1 = 0, v2 = 0, v3 = 0;
  if (base + 0 < n) v0 = in[base + 0];
  if (base + 1 < n) v1 = in[base + 1];
  if (base + 2 < n) v2 = in[base + 2];
  if (base + 3 < n) v3 = in[base + 3];
  int sum = v0 + v1 + v2 + v3;
  s[t] = sum;
  __syncthreads();
  for (int off = 1; off < 256; off <<= 1) {
    int tv = (t >= off) ? s[t - off] : 0;
    __syncthreads();
    s[t] += tv;
    __syncthreads();
  }
  int excl = s[t] - sum;
  if (t == 255) bsums[blockIdx.x] = s[255];
  if (base + 0 < n) out[base + 0] = excl;
  if (base + 1 < n) out[base + 1] = excl + v0;
  if (base + 2 < n) out[base + 2] = excl + v0 + v1;
  if (base + 3 < n) out[base + 3] = excl + v0 + v1 + v2;
}

__global__ void k_scan_sums(int* __restrict__ bsums, int nb) {
  __shared__ int s[1024];
  int t = threadIdx.x;
  int v = (t < nb) ? bsums[t] : 0;
  s[t] = v;
  __syncthreads();
  for (int off = 1; off < 1024; off <<= 1) {
    int tv = (t >= off) ? s[t - off] : 0;
    __syncthreads();
    s[t] += tv;
    __syncthreads();
  }
  if (t < nb) bsums[t] = s[t] - v;   // exclusive
}

__global__ void k_scan_add(int* __restrict__ out, const int* __restrict__ bsums, int n) {
  int i = blockIdx.x * 256 + threadIdx.x;
  if (i < n) out[i] += bsums[i >> 10];
}

// ---------------------------------------------------------------------------
// Host-side orchestration
// ---------------------------------------------------------------------------
extern "C" void kernel_launch(void* const* d_in, const int* in_sizes, int n_in,
                              void* d_out, int out_size, void* d_ws, size_t ws_size,
                              hipStream_t stream) {
  (void)n_in; (void)out_size; (void)ws_size;
  constexpr int C = 128;
  constexpr int Bg = 256;              // num_graphs (reference constant)
  const int N = in_sizes[2];           // 100352 nodes
  const int E = in_sizes[1] / 2;       // 1048576 edges (power of two)
  const int num_new = N - Bg;          // 100096
  const int epg = E / Bg;              // 4096 edges per graph (contiguous)

  const float* x    = (const float*)d_in[0];
  const int*   ei   = (const int*)d_in[1];
  const int*   src  = ei;
  const int*   dst  = ei + E;
  const int*   batch= (const int*)d_in[2];
  const float* W1   = (const float*)d_in[4];
  const float* b1   = (const float*)d_in[5];
  const float* W2   = (const float*)d_in[6];
  const float* b2   = (const float*)d_in[7];
  const float* Wm1  = (const float*)d_in[8];
  const float* bm1  = (const float*)d_in[9];
  const float* Wm2  = (const float*)d_in[10];
  const float* bm2  = (const float*)d_in[11];

  float* out_x   = (float*)d_out;
  int*   out_e0  = (int*)(out_x + (size_t)num_new * C);
  int*   out_e1  = out_e0 + E;
  int*   out_nb  = out_e1 + E;

  // workspace carve (256B aligned)
  char* wp = (char*)d_ws;
  auto carve = [&](size_t bytes) {
    char* p = wp;
    wp += (bytes + 255) & ~(size_t)255;
    return p;
  };
  unsigned short* xh    = (unsigned short*)carve((size_t)N * C * 2);
  unsigned short* W1t   = (unsigned short*)carve((size_t)256 * 256 * 2);
  unsigned short* Wm1t  = (unsigned short*)carve((size_t)C * C * 2);
  unsigned short* Wm2t  = (unsigned short*)carve((size_t)C * C * 2);
  float*          scores= (float*)carve((size_t)E * 4);
  float*          xm    = (float*)carve((size_t)N * C * 4);
  int*            sel_s = (int*)carve((size_t)Bg * 4);
  int*            sel_d = (int*)carve((size_t)Bg * 4);
  int*            flag  = (int*)carve((size_t)N * 4);
  int*            pos   = (int*)carve((size_t)N * 4);
  int*            remix = (int*)carve((size_t)(N - 2 * Bg) * 4);
  int*            clus  = (int*)carve((size_t)N * 4);
  unsigned short* tmpbf = (unsigned short*)carve((size_t)num_new * C * 2);
  unsigned long long* keys = (unsigned long long*)carve((size_t)E * 8);
  int*            uf    = (int*)carve((size_t)E * 4);
  int*            upos  = (int*)carve((size_t)E * 4);
  int*            bsums = (int*)carve((size_t)1024 * 4);

  auto cdiv = [](long long a, long long b) { return (int)((a + b - 1) / b); };

  // 1) precision conversion / weight transposes
  k_cvt_bf16<<<cdiv((size_t)N * C, 256), 256, 0, stream>>>(x, xh, N * C);
  k_transpose_bf16<<<cdiv(256 * 256, 256), 256, 0, stream>>>(W1, W1t, 256, 256);
  k_transpose_bf16<<<cdiv(C * C, 256), 256, 0, stream>>>(Wm1, Wm1t, C, C);
  k_transpose_bf16<<<cdiv(C * C, 256), 256, 0, stream>>>(Wm2, Wm2t, C, C);

  // 2) edge MLP (WMMA) -> scores
  k_scores_init<<<cdiv(E, 256), 256, 0, stream>>>(scores, b2, E);
  {
    size_t smem = (size_t)128 * A_STRIDE * 2 + (size_t)256 * W_STRIDE * 2; // 202752
    k_edge_mlp<<<E / 128, 256, smem, stream>>>(xh, W1t, b1, W2, src, dst, scores);
  }

  // 3) mlp1 on nodes (WMMA): xm = x @ Wm1 + bm1
  {
    size_t smem = (size_t)2 * 128 * NA_STRIDE * 2; // 69632
    k_gemm128<<<N / 128, 256, smem, stream>>>(xh, Wm1t, bm1, xm);
  }

  // 4) per-graph argmax edge
  k_argmax<<<Bg, 256, 0, stream>>>(scores, src, dst, sel_s, sel_d, epg);

  // 5) mask + cluster
  k_fill_int<<<cdiv(N, 256), 256, 0, stream>>>(flag, 1, N);
  k_mark<<<1, 256, 0, stream>>>(sel_s, sel_d, flag, clus, Bg);
  {
    int nb = cdiv(N, 1024);
    k_scan_block<<<nb, 256, 0, stream>>>(flag, pos, bsums, N);
    k_scan_sums<<<1, 1024, 0, stream>>>(bsums, nb);
    k_scan_add<<<cdiv(N, 256), 256, 0, stream>>>(pos, bsums, N);
  }
  k_remidx<<<cdiv(N, 256), 256, 0, stream>>>(flag, pos, remix, clus, N, Bg);

  // 6) build merged/remaining rows, mlp2 (WMMA) -> new_x
  k_build_tmp<<<cdiv((size_t)num_new * C, 256), 256, 0, stream>>>(
      xm, sel_s, sel_d, remix, tmpbf, num_new, Bg);
  {
    size_t smem = (size_t)2 * 128 * NA_STRIDE * 2;
    k_gemm128<<<num_new / 128, 256, smem, stream>>>(tmpbf, Wm2t, bm2, out_x);
  }
  k_new_batch<<<cdiv(num_new, 256), 256, 0, stream>>>(batch, remix, out_nb, num_new, Bg);

  // 7) coalesce mapped edges: sort + unique, pad with (num_new-1,num_new-1)
  k_keys<<<cdiv(E, 256), 256, 0, stream>>>(src, dst, clus, keys, E);
  for (int k = 2; k <= E; k <<= 1)
    for (int j = k >> 1; j > 0; j >>= 1)
      k_bitonic<<<E / 256, 256, 0, stream>>>(keys, j, k);
  k_uniq<<<cdiv(E, 256), 256, 0, stream>>>(keys, uf, E);
  {
    int nb = cdiv(E, 1024);   // 1024 blocks for E=2^20
    k_scan_block<<<nb, 256, 0, stream>>>(uf, upos, bsums, E);
    k_scan_sums<<<1, 1024, 0, stream>>>(bsums, nb);
    k_scan_add<<<cdiv(E, 256), 256, 0, stream>>>(upos, bsums, E);
  }
  k_fill_int<<<cdiv(E, 256), 256, 0, stream>>>(out_e0, num_new - 1, E);
  k_fill_int<<<cdiv(E, 256), 256, 0, stream>>>(out_e1, num_new - 1, E);
  k_scatter_unique<<<cdiv(E, 256), 256, 0, stream>>>(keys, uf, upos, out_e0, out_e1, E);
}
